// MoleculeGCN_50165218017611
// MI455X (gfx1250) — matrix operand order repaired and checked by reading, not compiled, not run on hardware
//
#include <hip/hip_runtime.h>
#include <hip/hip_bf16.h>

#define NN 50000          // nodes
#define NE 1250000        // edges
#define NG 256            // graphs
#define SP0 0.6931471805599453f   // softplus(0)

typedef __attribute__((ext_vector_type(8)))  float  v8f;
typedef __attribute__((ext_vector_type(16))) __bf16 v16bf;

union Frag { unsigned u[8]; v16bf v; };

__device__ __forceinline__ float sp(float x) {
    return fmaxf(x, 0.f) + log1pf(expf(-fabsf(x)));
}
__device__ __forceinline__ unsigned short f2bf(float f) {
    unsigned u = __float_as_uint(f);
    u += 0x7FFFu + ((u >> 16) & 1u);           // round-to-nearest-even
    return (unsigned short)(u >> 16);
}
// order-preserving float<->u32 encoding for atomicMax-based segment max
__device__ __forceinline__ unsigned encmax(float f) {
    unsigned u = __float_as_uint(f);
    return (u & 0x80000000u) ? ~u : (u | 0x80000000u);
}
__device__ __forceinline__ float decmax(unsigned u) {
    return __uint_as_float((u & 0x80000000u) ? (u & 0x7FFFFFFFu) : ~u);
}

// ---- embeddings + softplus --------------------------------------------------
__global__ void k_init(const int* __restrict__ x, const int* __restrict__ eattr,
                       const float* __restrict__ embA, const float* __restrict__ embB,
                       float* __restrict__ atom, float* __restrict__ Bfeat) {
    int i = blockIdx.x * blockDim.x + threadIdx.x;
    if (i >= NN * 64) return;
    int r = i >> 6, f = i & 63;
    atom[i]  = sp(embA[x[r] * 64 + f]);
    Bfeat[i] = sp(embB[eattr[r] * 64 + f]);   // only bond rows < NN are ever gathered
}

// ---- CSR build (once; dst is iteration-invariant) ---------------------------
__global__ void k_indeg(const int* __restrict__ dst, unsigned* __restrict__ indeg) {
    int e = blockIdx.x * blockDim.x + threadIdx.x;
    if (e < NE) atomicAdd(&indeg[dst[e]], 1u);
}

__global__ void k_scan(const unsigned* __restrict__ indeg, unsigned* __restrict__ off) {
    __shared__ unsigned part[1024];
    int t = threadIdx.x;
    const int CH = (NN + 1023) / 1024;
    unsigned s = 0;
    for (int i = 0; i < CH; ++i) { int idx = t * CH + i; if (idx < NN) s += indeg[idx]; }
    part[t] = s; __syncthreads();
    for (int d = 1; d < 1024; d <<= 1) {
        unsigned v = (t >= d) ? part[t - d] : 0u;
        __syncthreads(); part[t] += v; __syncthreads();
    }
    unsigned run = (t > 0) ? part[t - 1] : 0u;
    for (int i = 0; i < CH; ++i) {
        int idx = t * CH + i;
        if (idx < NN) { off[idx] = run; run += indeg[idx]; }
    }
    if (t == 1023) off[NN] = part[1023];
}

__global__ void k_fillcsr(const int* __restrict__ dst, const unsigned* __restrict__ off,
                          unsigned* __restrict__ cursor, unsigned* __restrict__ csr) {
    int e = blockIdx.x * blockDim.x + threadIdx.x;
    if (e >= NE) return;
    int d = dst[e];
    unsigned p = off[d] + atomicAdd(&cursor[d], 1u);
    csr[p] = (unsigned)e;
}

// ---- EdgeConv: 16-edge tiles, concat(h_i, h_j-h_i) @ W_e via bf16 WMMA ------
// A: 16x128 (M=edges, K=feat), B: W_e chunks 32x64, 4 K-chunks x 4 N-tiles = 16 WMMA/tile.
// dst-sorted order -> run-compressed encoded atomicMax scatter.
__global__ __launch_bounds__(256) void k_edgeconv(
    const float* __restrict__ Bfeat, const int* __restrict__ src, const int* __restrict__ dst,
    const unsigned* __restrict__ csr, const float* __restrict__ W_e,
    const float* __restrict__ b_e, unsigned* __restrict__ agg)
{
    __shared__ __align__(16) unsigned short Wt[64][128];     // Wt[n][k] = bf16(W_e[k][n])
    __shared__ __align__(16) unsigned short As[8][16][128];  // per-wave A tile
    __shared__ int dstIds[8][16];

    for (int i = threadIdx.x; i < 128 * 64; i += 256) {
        int k = i >> 6, n = i & 63;
        Wt[n][k] = f2bf(W_e[i]);
    }
    __syncthreads();

    const int wave = threadIdx.x >> 5, lane = threadIdx.x & 31;
    const int l15 = lane & 15, hi16 = lane >> 4;

    // B fragments held in registers for the whole grid-stride loop
    Frag bw[4][4];
    #pragma unroll
    for (int n = 0; n < 4; ++n)
        #pragma unroll
        for (int kc = 0; kc < 4; ++kc)
            #pragma unroll
            for (int u = 0; u < 8; ++u) {
                int K = 32 * kc + 2 * u + ((u >= 4) ? 8 : 0) + hi16 * 8;  // ISA 16-bit B layout
                bw[n][kc].u[u] = *(const unsigned*)&Wt[16 * n + l15][K];
            }
    float bias[4];
    #pragma unroll
    for (int n = 0; n < 4; ++n) bias[n] = b_e[16 * n + l15];

    const int nTiles = NE / 16;  // 78125
    for (int base = blockIdx.x * 8; base < nTiles; base += gridDim.x * 8) {
        int t = base + wave;
        bool valid = t < nTiles;
        if (valid) {
            int m = lane >> 1;
            int e = (int)csr[t * 16 + m];
            int d = dst[e], s = src[e];
            if ((lane & 1) == 0) dstIds[wave][m] = d;
            const float* hr = Bfeat + (size_t)d * 64;
            const float* jr = Bfeat + (size_t)s * 64;
            int f0 = (lane & 1) * 32;
            #pragma unroll 4
            for (int j = 0; j < 32; ++j) {
                int f = f0 + j;
                float a = hr[f], b = jr[f];
                As[wave][m][f]      = f2bf(a);       // h_i
                As[wave][m][64 + f] = f2bf(b - a);   // h_j - h_i
            }
        }
        __syncthreads();
        if (valid) {
            v8f acc[4];
            #pragma unroll
            for (int n = 0; n < 4; ++n)
                #pragma unroll
                for (int q = 0; q < 8; ++q) acc[n][q] = 0.f;
            #pragma unroll
            for (int kc = 0; kc < 4; ++kc) {
                Frag af;
                #pragma unroll
                for (int u = 0; u < 8; ++u) {
                    int K = 32 * kc + 2 * u + ((u >= 4) ? 8 : 0) + hi16 * 8;  // ISA 16-bit A layout
                    af.u[u] = *(const unsigned*)&As[wave][l15][K];
                }
                #pragma unroll
                for (int n = 0; n < 4; ++n)
                    acc[n] = __builtin_amdgcn_wmma_f32_16x16x32_bf16(
                        false, af.v, false, bw[n][kc].v, (short)0, acc[n], false, false);
            }
            // scatter-max, run-compressed over consecutive same-dst rows
            #pragma unroll
            for (int n = 0; n < 4; ++n) {
                int N = 16 * n + l15;
                int rund = dstIds[wave][8 * hi16];
                float runmax = acc[n][0] + bias[n];
                #pragma unroll
                for (int q = 1; q < 8; ++q) {
                    int dd = dstIds[wave][q + 8 * hi16];
                    float val = acc[n][q] + bias[n];
                    if (dd != rund) {
                        atomicMax(&agg[(size_t)rund * 64 + N], encmax(runmax));
                        rund = dd; runmax = val;
                    } else runmax = fmaxf(runmax, val);
                }
                atomicMax(&agg[(size_t)rund * 64 + N], encmax(runmax));
            }
        }
        __syncthreads();
    }
}

// ---- out_bond = sp(where(has_in, agg, 0)) (active rows only) ----------------
__global__ void k_bondupd(const unsigned* __restrict__ agg, const unsigned* __restrict__ indeg,
                          float* __restrict__ Bfeat) {
    int i = blockIdx.x * blockDim.x + threadIdx.x;
    if (i >= NN * 64) return;
    int r = i >> 6;
    Bfeat[i] = (indeg[r] > 0u) ? sp(decmax(agg[i])) : SP0;
}

// ---- GeneralConv gather: S[i]=sum atom[src], T[i]=sum bond_pe[e] (atomic-free)
__global__ void k_gather_st(const float* __restrict__ atom, const float* __restrict__ Bfeat,
                            const int* __restrict__ src, const unsigned* __restrict__ csr,
                            const unsigned* __restrict__ off,
                            float* __restrict__ S, float* __restrict__ T) {
    int node = blockIdx.x * 8 + (threadIdx.x >> 5);
    if (node >= NN) return;
    int lane = threadIdx.x & 31;
    unsigned p0 = off[node], p1 = off[node + 1];
    float s0 = 0, s1 = 0, t0 = 0, t1 = 0; unsigned chi = 0;
    for (unsigned p = p0; p < p1; ++p) {
        int e = (int)csr[p];
        const float* ar = atom + (size_t)src[e] * 64;
        s0 += ar[lane]; s1 += ar[lane + 32];
        if (e < NN) {
            const float* br = Bfeat + (size_t)e * 64;
            t0 += br[lane]; t1 += br[lane + 32];
        } else ++chi;                        // bond rows >= NN are softplus(0)
    }
    t0 += (float)chi * SP0; t1 += (float)chi * SP0;
    size_t b = (size_t)node * 64;
    S[b + lane] = s0; S[b + lane + 32] = s1;
    T[b + lane] = t0; T[b + lane + 32] = t1;
}

// ---- GeneralConv node update via WMMA: atom = sp(S@Wm + T@We + indeg*(bm+be) + atom)
__global__ __launch_bounds__(256) void k_nodeupd(
    const float* __restrict__ S, const float* __restrict__ T, const unsigned* __restrict__ indeg,
    const float* __restrict__ W_msg, const float* __restrict__ b_msg,
    const float* __restrict__ W_edge, const float* __restrict__ b_edge,
    float* __restrict__ atom)
{
    __shared__ __align__(16) unsigned short Wm[64][64], We[64][64];
    __shared__ __align__(16) unsigned short A1[8][16][64], A2[8][16][64];
    for (int i = threadIdx.x; i < 64 * 64; i += 256) {
        int k = i >> 6, n = i & 63;
        Wm[n][k] = f2bf(W_msg[i]);
        We[n][k] = f2bf(W_edge[i]);
    }
    __syncthreads();
    const int wave = threadIdx.x >> 5, lane = threadIdx.x & 31;
    const int l15 = lane & 15, hi16 = lane >> 4;
    Frag bm[4][2], be[4][2];
    #pragma unroll
    for (int n = 0; n < 4; ++n)
        #pragma unroll
        for (int kc = 0; kc < 2; ++kc)
            #pragma unroll
            for (int u = 0; u < 8; ++u) {
                int K = 32 * kc + 2 * u + ((u >= 4) ? 8 : 0) + hi16 * 8;
                bm[n][kc].u[u] = *(const unsigned*)&Wm[16 * n + l15][K];
                be[n][kc].u[u] = *(const unsigned*)&We[16 * n + l15][K];
            }
    float bsum[4];
    #pragma unroll
    for (int n = 0; n < 4; ++n) bsum[n] = b_msg[16 * n + l15] + b_edge[16 * n + l15];

    const int nTiles = NN / 16;   // 3125
    int t = blockIdx.x * 8 + wave;
    bool valid = t < nTiles;
    if (valid) {
        int m = lane >> 1, f0 = (lane & 1) * 32;
        size_t rb = (size_t)(t * 16 + m) * 64;
        #pragma unroll 4
        for (int j = 0; j < 32; ++j) {
            int f = f0 + j;
            A1[wave][m][f] = f2bf(S[rb + f]);
            A2[wave][m][f] = f2bf(T[rb + f]);
        }
    }
    __syncthreads();
    if (!valid) return;
    v8f acc[4];
    #pragma unroll
    for (int n = 0; n < 4; ++n)
        #pragma unroll
        for (int q = 0; q < 8; ++q) acc[n][q] = 0.f;
    #pragma unroll
    for (int kc = 0; kc < 2; ++kc) {
        Frag a1, a2;
        #pragma unroll
        for (int u = 0; u < 8; ++u) {
            int K = 32 * kc + 2 * u + ((u >= 4) ? 8 : 0) + hi16 * 8;
            a1.u[u] = *(const unsigned*)&A1[wave][l15][K];
            a2.u[u] = *(const unsigned*)&A2[wave][l15][K];
        }
        #pragma unroll
        for (int n = 0; n < 4; ++n) {
            acc[n] = __builtin_amdgcn_wmma_f32_16x16x32_bf16(
                false, a1.v, false, bm[n][kc].v, (short)0, acc[n], false, false);
            acc[n] = __builtin_amdgcn_wmma_f32_16x16x32_bf16(
                false, a2.v, false, be[n][kc].v, (short)0, acc[n], false, false);
        }
    }
    #pragma unroll
    for (int n = 0; n < 4; ++n) {
        int N = 16 * n + l15;
        #pragma unroll
        for (int q = 0; q < 8; ++q) {
            int row = t * 16 + q + 8 * hi16;
            size_t idx = (size_t)row * 64 + N;
            float v = acc[n][q] + (float)indeg[row] * bsum[n] + atom[idx];
            atom[idx] = sp(v);
        }
    }
}

// ---- readout ----------------------------------------------------------------
__global__ void k_pool(const float* __restrict__ atom, const int* __restrict__ batch,
                       float* __restrict__ G) {
    int i = blockIdx.x * blockDim.x + threadIdx.x;
    if (i >= NN * 64) return;
    int r = i >> 6, f = i & 63;
    atomicAdd(&G[batch[r] * 64 + f], atom[i]);
}

__global__ __launch_bounds__(256) void k_mlp(
    const float* __restrict__ G,
    const float* __restrict__ Wr0, const float* __restrict__ br0,
    const float* __restrict__ Wr1, const float* __restrict__ br1,
    const float* __restrict__ Wout, const float* __restrict__ bout,
    float* __restrict__ out)
{
    __shared__ float sW0[64 * 64], sW1[64 * 64], sb0[64], sb1[64], sWo[64];
    for (int i = threadIdx.x; i < 64 * 64; i += 256) { sW0[i] = Wr0[i]; sW1[i] = Wr1[i]; }
    if (threadIdx.x < 64) {
        sb0[threadIdx.x] = br0[threadIdx.x];
        sb1[threadIdx.x] = br1[threadIdx.x];
        sWo[threadIdx.x] = Wout[threadIdx.x];
    }
    __syncthreads();
    int g = threadIdx.x;   // one thread per graph, 256 total
    float h0[64], h1[64];
    for (int o = 0; o < 64; ++o) {
        float a = sb0[o];
        for (int k = 0; k < 64; ++k) a += G[g * 64 + k] * sW0[k * 64 + o];
        h0[o] = sp(a);
    }
    for (int o = 0; o < 64; ++o) {
        float a = sb1[o];
        for (int k = 0; k < 64; ++k) a += h0[k] * sW1[k * 64 + o];
        h1[o] = sp(a);
    }
    float a = bout[0];
    for (int k = 0; k < 64; ++k) a += h1[k] * sWo[k];
    out[g] = a;
}

// ---- expand final out_bond to full [NE,64] (rows >= NN are softplus(0)) -----
__global__ void k_writebond(const float* __restrict__ Bfeat, float* __restrict__ bond) {
    const long long total4 = (long long)NE * 16;   // NE*64/4
    long long i = (long long)blockIdx.x * blockDim.x + threadIdx.x;
    long long stride = (long long)gridDim.x * blockDim.x;
    const float4 c = make_float4(SP0, SP0, SP0, SP0);
    for (; i < total4; i += stride) {
        long long base = i * 4;
        float4 v = (base < (long long)NN * 64) ? *(const float4*)(Bfeat + base) : c;
        ((float4*)bond)[i] = v;
    }
}

extern "C" void kernel_launch(void* const* d_in, const int* in_sizes, int n_in,
                              void* d_out, int out_size, void* d_ws, size_t ws_size,
                              hipStream_t stream)
{
    (void)in_sizes; (void)n_in; (void)out_size; (void)ws_size;
    const int*   x     = (const int*)d_in[0];
    const int*   eattr = (const int*)d_in[1];
    const int*   eidx  = (const int*)d_in[2];
    const int*   batch = (const int*)d_in[3];
    const float* embA  = (const float*)d_in[4];
    const float* embB  = (const float*)d_in[5];
    const float* W_e   = (const float*)d_in[6];
    const float* b_e   = (const float*)d_in[7];
    const float* Wmsg  = (const float*)d_in[8];
    const float* bmsg  = (const float*)d_in[9];
    const float* Wedg  = (const float*)d_in[10];
    const float* bedg  = (const float*)d_in[11];
    const float* Wr0   = (const float*)d_in[12];
    const float* br0   = (const float*)d_in[13];
    const float* Wr1   = (const float*)d_in[14];
    const float* br1   = (const float*)d_in[15];
    const float* Wout  = (const float*)d_in[16];
    const float* bout  = (const float*)d_in[17];
    const int* src  = eidx;
    const int* dstp = eidx + NE;

    float* out  = (float*)d_out;
    float* atom = out + NG;                    // out_atom lives in d_out, updated in place
    float* bond = atom + (size_t)NN * 64;

    char* w = (char*)d_ws;
    float*    Bfeat  = (float*)w;    w += (size_t)NN * 64 * 4;
    unsigned* agg    = (unsigned*)w; w += (size_t)NN * 64 * 4;
    float*    S      = (float*)w;    w += (size_t)NN * 64 * 4;
    float*    Tb     = (float*)w;    w += (size_t)NN * 64 * 4;
    unsigned* indeg  = (unsigned*)w; w += (size_t)NN * 4;
    unsigned* off    = (unsigned*)w; w += (size_t)(NN + 1) * 4;
    unsigned* cursor = (unsigned*)w; w += (size_t)NN * 4;
    unsigned* csr    = (unsigned*)w; w += (size_t)NE * 4;
    float*    G      = (float*)w;    w += (size_t)NG * 64 * 4;

    hipMemsetAsync(indeg,  0, (size_t)NN * 4, stream);
    hipMemsetAsync(cursor, 0, (size_t)NN * 4, stream);
    hipMemsetAsync(G,      0, (size_t)NG * 64 * 4, stream);

    k_init<<<(NN * 64 + 255) / 256, 256, 0, stream>>>(x, eattr, embA, embB, atom, Bfeat);
    k_indeg<<<(NE + 255) / 256, 256, 0, stream>>>(dstp, indeg);
    k_scan<<<1, 1024, 0, stream>>>(indeg, off);
    k_fillcsr<<<(NE + 255) / 256, 256, 0, stream>>>(dstp, off, cursor, csr);

    for (int it = 0; it < 2; ++it) {
        hipMemsetAsync(agg, 0, (size_t)NN * 64 * 4, stream);   // 0 == encoded -inf floor
        k_edgeconv<<<2048, 256, 0, stream>>>(Bfeat, src, dstp, csr, W_e, b_e, agg);
        k_bondupd<<<(NN * 64 + 255) / 256, 256, 0, stream>>>(agg, indeg, Bfeat);
        k_gather_st<<<NN / 8, 256, 0, stream>>>(atom, Bfeat, src, csr, off, S, Tb);
        k_nodeupd<<<(NN / 16 + 7) / 8, 256, 0, stream>>>(S, Tb, indeg, Wmsg, bmsg, Wedg, bedg, atom);
    }

    k_pool<<<(NN * 64 + 255) / 256, 256, 0, stream>>>(atom, batch, G);
    k_mlp<<<1, 256, 0, stream>>>(G, Wr0, br0, Wr1, br1, Wout, bout, out);
    k_writebond<<<4096, 256, 0, stream>>>(Bfeat, bond);
}
